// MoELayer_76304388981206
// MI455X (gfx1250) — matrix operand order repaired and checked by reading, not compiled, not run on hardware
//
#include <hip/hip_runtime.h>
#include <math.h>
#include <stdint.h>

// ---------------- problem constants ----------------
constexpr int kH = 1024;      // hidden
constexpr int kI = 4096;      // intermediate
constexpr int kE = 8;         // experts
constexpr int kT = 4096;      // tokens (2*2048)
constexpr int kPairs = kT * 2;
constexpr float kLB = 0.01f;

// ---------------- GEMM tiling ----------------
constexpr int BM = 64;
constexpr int BN = 128;
constexpr int BK = 32;
constexpr int LDA = BK + 8;   // bf16 elems, 80B row pitch (16B aligned)
constexpr int LDB = BK + 8;   // B tile: BN rows of BK k-values (pre-transposed weights)

typedef __attribute__((ext_vector_type(16))) __bf16 v16bf;
typedef __attribute__((ext_vector_type(8)))  __bf16 bf16x8;
typedef __attribute__((ext_vector_type(8)))  float  v8f;

static __device__ __forceinline__ __bf16 f2bf(float f) {
    unsigned u = __builtin_bit_cast(unsigned, f);
    unsigned r = (u + 0x7FFFu + ((u >> 16) & 1u)) >> 16;   // RNE
    unsigned short s = (unsigned short)r;
    return __builtin_bit_cast(__bf16, s);
}
static __device__ __forceinline__ float gelu_tanh(float v) {
    const float c = 0.7978845608028654f;
    float u = c * (v + 0.044715f * v * v * v);
    return 0.5f * v * (1.0f + tanhf(u));
}

// ---- gfx1250 async copy: global -> LDS, 16B per lane, ASYNCcnt tracked ----
static __device__ __forceinline__ void async_ld_b128(unsigned lds_off, const void* g) {
    unsigned long long ga = (unsigned long long)(uintptr_t)g;
    asm volatile("global_load_async_to_lds_b128 %0, %1, off"
                 :: "v"(lds_off), "v"(ga) : "memory");
}
static __device__ __forceinline__ void wait_async0() {
    asm volatile("s_wait_asynccnt 0x0" ::: "memory");
}
static __device__ __forceinline__ unsigned lds_off32(const void* p) {
    return (unsigned)(uintptr_t)p;   // LDS aperture: addr[31:0] == LDS byte offset
}

// ================= kernel 0: zero out + meta =================
__global__ void k_zero(float* out, int nout, int* meta /*25 ints*/, float* probs /*8 f*/) {
    int i = blockIdx.x * blockDim.x + threadIdx.x;
    if (i < nout) out[i] = 0.0f;
    if (i < 25) meta[i] = 0;
    else if (i < 33) probs[i - 25] = 0.0f;
}

// ================= kernel 1: fp32 -> bf16 (activations) =================
__global__ void k_cvt(const float* __restrict__ x, __bf16* __restrict__ xb, int n) {
    int i = blockIdx.x * blockDim.x + threadIdx.x;
    if (i < n) xb[i] = f2bf(x[i]);
}

// ====== kernel 1b: per-expert transpose W[e][K][N] fp32 -> Wt[e][N][K] bf16 ======
__global__ void k_transpose(const float* __restrict__ src, __bf16* __restrict__ dst,
                            int K, int N) {
    __shared__ float tile[32][33];
    const int e = blockIdx.z;
    const int n0 = blockIdx.x * 32;
    const int k0 = blockIdx.y * 32;
    const int tx = threadIdx.x;       // 32
    const int ty = threadIdx.y;       // 8
    const float* s = src + (size_t)e * K * N;
    __bf16* d = dst + (size_t)e * N * K;
#pragma unroll
    for (int j = 0; j < 4; ++j)
        tile[ty + j * 8][tx] = s[(size_t)(k0 + ty + j * 8) * N + n0 + tx];
    __syncthreads();
#pragma unroll
    for (int j = 0; j < 4; ++j)
        d[(size_t)(n0 + ty + j * 8) * K + k0 + tx] = f2bf(tile[tx][ty + j * 8]);
}

// ================= kernel 2: router =================
__global__ void k_router(const float* __restrict__ x, const float* __restrict__ Wr,
                         const float* __restrict__ br,
                         int* __restrict__ counts, float* __restrict__ probs_sum,
                         int* __restrict__ top2i, float* __restrict__ top2w) {
    int t = blockIdx.x * blockDim.x + threadIdx.x;
    if (t >= kT) return;

    float lg[kE];
#pragma unroll
    for (int e = 0; e < kE; ++e) lg[e] = br[e];
    const float* xr = x + (size_t)t * kH;
    for (int h = 0; h < kH; ++h) {
        float xv = xr[h];
#pragma unroll
        for (int e = 0; e < kE; ++e) lg[e] += xv * Wr[h * kE + e];
    }
    float m = lg[0];
#pragma unroll
    for (int e = 1; e < kE; ++e) m = fmaxf(m, lg[e]);
    float p[kE]; float s = 0.0f;
#pragma unroll
    for (int e = 0; e < kE; ++e) { p[e] = expf(lg[e] - m); s += p[e]; }
    float inv = 1.0f / s;
#pragma unroll
    for (int e = 0; e < kE; ++e) { p[e] *= inv; atomicAdd(&probs_sum[e], p[e]); }

    int i0 = 0;
#pragma unroll
    for (int e = 1; e < kE; ++e) if (p[e] > p[i0]) i0 = e;
    int i1 = (i0 == 0) ? 1 : 0;
#pragma unroll
    for (int e = 0; e < kE; ++e) if (e != i0 && p[e] > p[i1]) i1 = e;

    float w0 = p[i0], w1 = p[i1];
    float rs = 1.0f / (w0 + w1);
    top2i[t * 2 + 0] = i0;  top2w[t * 2 + 0] = w0 * rs;
    top2i[t * 2 + 1] = i1;  top2w[t * 2 + 1] = w1 * rs;
    atomicAdd(&counts[i0], 1);
    atomicAdd(&counts[i1], 1);
}

// ================= kernel 3: offsets + aux loss =================
__global__ void k_finalize(const int* __restrict__ counts, const float* __restrict__ probs_sum,
                           int* __restrict__ offsets, float* __restrict__ aux_out) {
    if (threadIdx.x != 0 || blockIdx.x != 0) return;
    int acc = 0;
    for (int e = 0; e < kE; ++e) { offsets[e] = acc; acc += counts[e]; }
    offsets[kE] = acc;
    float aux = 0.0f;
    for (int e = 0; e < kE; ++e) {
        float ft = (float)counts[e] / (float)(kT * 2);
        float fp = probs_sum[e] / (float)kT;
        aux += ft * fp;
    }
    aux_out[0] = (float)kE * aux * kLB;
}

// ================= kernel 4: fill expert-grouped pair lists =================
__global__ void k_fill(const int* __restrict__ top2i, const float* __restrict__ top2w,
                       const int* __restrict__ offsets, int* __restrict__ cursor,
                       int* __restrict__ pair_token, float* __restrict__ pair_weight) {
    int t = blockIdx.x * blockDim.x + threadIdx.x;
    if (t >= kT) return;
#pragma unroll
    for (int k = 0; k < 2; ++k) {
        int e = top2i[t * 2 + k];
        int pos = atomicAdd(&cursor[e], 1);
        int g = offsets[e] + pos;
        pair_token[g] = t;
        pair_weight[g] = top2w[t * 2 + k];
    }
}

// ---------- fragment helpers (documented CDNA5 WMMA VGPR layouts) ----------
static __device__ __forceinline__ v16bf load_a_frag(const __bf16* lA, int lane, int wm) {
    int row = wm * 16 + (lane & 15);
    int koff = (lane & 16) ? 8 : 0;                 // lanes 16-31: K 8..15 / 24..31
    const bf16x8 p0 = *(const bf16x8*)&lA[row * LDA + koff];
    const bf16x8 p1 = *(const bf16x8*)&lA[row * LDA + 16 + koff];
    return __builtin_shufflevector(p0, p1, 0,1,2,3,4,5,6,7,8,9,10,11,12,13,14,15);
}
static __device__ __forceinline__ v16bf load_b_frag(const __bf16* lBt, int lane, int ncol) {
    int bk = (lane & 16) ? 8 : 0;                   // lanes 16-31: K 8..15 / 24..31
    const bf16x8 q0 = *(const bf16x8*)&lBt[ncol * LDB + bk];
    const bf16x8 q1 = *(const bf16x8*)&lBt[ncol * LDB + 16 + bk];
    return __builtin_shufflevector(q0, q1, 0,1,2,3,4,5,6,7,8,9,10,11,12,13,14,15);
}

// ---------- staging (async DMA to LDS, per-lane 16B) ----------
// A tile: 64 rows x 32 bf16 (64B/row) = 256 x 16B : thread -> (row=tid>>2, seg=tid&3)
// B tile: 128 rows x 32 bf16 (64B/row) = 512 x 16B: thread -> (row=tid>>1, 2 segs)
static __device__ __forceinline__ void stage_a_gather(__bf16* lA, const __bf16* Xbf,
                                                      const int* pair_token, int base,
                                                      int m0, int cnt, int k0, int tid) {
    int row = tid >> 2, seg = tid & 3;
    int mg = m0 + row;
    if (mg < cnt) {
        int tok = pair_token[base + mg];
        async_ld_b128(lds_off32(&lA[row * LDA + seg * 8]),
                      Xbf + (size_t)tok * kH + k0 + seg * 8);
    }
}
static __device__ __forceinline__ void stage_a_rows(__bf16* lA, const __bf16* Hbuf,
                                                    int base, int m0, int cnt, int k0, int tid) {
    int row = tid >> 2, seg = tid & 3;
    int mg = m0 + row;
    if (mg < cnt)
        async_ld_b128(lds_off32(&lA[row * LDA + seg * 8]),
                      Hbuf + (size_t)(base + mg) * kI + k0 + seg * 8);
}
static __device__ __forceinline__ void stage_b(__bf16* lBt, const __bf16* Wt,
                                               size_t ebase, int Kdim, int n0, int k0, int tid) {
    int row = tid >> 1;
    int s0 = (tid & 1) * 2;
#pragma unroll
    for (int s = 0; s < 2; ++s)
        async_ld_b128(lds_off32(&lBt[row * LDB + (s0 + s) * 8]),
                      Wt + ebase + (size_t)(n0 + row) * Kdim + k0 + (s0 + s) * 8);
}

// ================= kernel 5: GEMM1  h = gelu(X W1 + b1) -> Hbuf (bf16) ======
__global__ __launch_bounds__(256) void
k_gemm1(const __bf16* __restrict__ Xbf, const __bf16* __restrict__ W1t,
        const float* __restrict__ b1, const int* __restrict__ pair_token,
        const int* __restrict__ counts, const int* __restrict__ offsets,
        __bf16* __restrict__ Hbuf) {
    const int e = blockIdx.z;
    const int cnt = counts[e];
    const int m0 = blockIdx.y * BM;
    if (m0 >= cnt) return;
    const int base = offsets[e];
    const int n0 = blockIdx.x * BN;
    const size_t ebase = (size_t)e * kI * kH;   // W1t: [E][I][H] (N-major, K=H contiguous)

    __shared__ __align__(16) __bf16 lA[2][BM * LDA];
    __shared__ __align__(16) __bf16 lBt[2][BN * LDB];

    const int tid = threadIdx.x;
    const int lane = tid & 31;
    const int wv = tid >> 5;
    const int wm = wv & 3;              // 16-row M subtile
    const int wn = wv >> 2;             // 64-col N half

    v8f acc[4];
#pragma unroll
    for (int t = 0; t < 4; ++t) acc[t] = (v8f){0,0,0,0,0,0,0,0};

    constexpr int NK = kH / BK;
    stage_a_gather(lA[0], Xbf, pair_token, base, m0, cnt, 0, tid);
    stage_b(lBt[0], W1t, ebase, kH, n0, 0, tid);
    wait_async0();
    __syncthreads();

    for (int kt = 0; kt < NK; ++kt) {
        const int cur = kt & 1;
        if (kt + 1 < NK) {
            stage_a_gather(lA[cur ^ 1], Xbf, pair_token, base, m0, cnt, (kt + 1) * BK, tid);
            stage_b(lBt[cur ^ 1], W1t, ebase, kH, n0, (kt + 1) * BK, tid);
        }
        v16bf a = load_a_frag(lA[cur], lane, wm);
#pragma unroll
        for (int t = 0; t < 4; ++t) {
            v16bf b = load_b_frag(lBt[cur], lane, wn * 64 + t * 16 + (lane & 15));
            acc[t] = __builtin_amdgcn_wmma_f32_16x16x32_bf16(false, a, false, b,
                                                             (short)0, acc[t], false, false);
        }
        wait_async0();
        __syncthreads();
    }

    const int mhi = (lane & 16) ? 8 : 0;
    const int nlow = lane & 15;
#pragma unroll
    for (int t = 0; t < 4; ++t) {
        int nc = n0 + wn * 64 + t * 16 + nlow;
        float bias = b1[(size_t)e * kI + nc];
#pragma unroll
        for (int r = 0; r < 8; ++r) {
            int mg = m0 + wm * 16 + r + mhi;
            if (mg < cnt)
                Hbuf[(size_t)(base + mg) * kI + nc] = f2bf(gelu_tanh(acc[t][r] + bias));
        }
    }
}

// ================= kernel 6: GEMM2  out += w * (H W2 + b2) ==================
__global__ __launch_bounds__(256) void
k_gemm2(const __bf16* __restrict__ Hbuf, const __bf16* __restrict__ W2t,
        const float* __restrict__ b2, const int* __restrict__ pair_token,
        const float* __restrict__ pair_weight, const int* __restrict__ counts,
        const int* __restrict__ offsets, float* __restrict__ out) {
    const int e = blockIdx.z;
    const int cnt = counts[e];
    const int m0 = blockIdx.y * BM;
    if (m0 >= cnt) return;
    const int base = offsets[e];
    const int n0 = blockIdx.x * BN;
    const size_t ebase = (size_t)e * kH * kI;   // W2t: [E][H][I] (N-major, K=I contiguous)

    __shared__ __align__(16) __bf16 lA[2][BM * LDA];
    __shared__ __align__(16) __bf16 lBt[2][BN * LDB];

    const int tid = threadIdx.x;
    const int lane = tid & 31;
    const int wv = tid >> 5;
    const int wm = wv & 3;
    const int wn = wv >> 2;

    v8f acc[4];
#pragma unroll
    for (int t = 0; t < 4; ++t) acc[t] = (v8f){0,0,0,0,0,0,0,0};

    constexpr int NK = kI / BK;
    stage_a_rows(lA[0], Hbuf, base, m0, cnt, 0, tid);
    stage_b(lBt[0], W2t, ebase, kI, n0, 0, tid);
    wait_async0();
    __syncthreads();

    for (int kt = 0; kt < NK; ++kt) {
        const int cur = kt & 1;
        if (kt + 1 < NK) {
            stage_a_rows(lA[cur ^ 1], Hbuf, base, m0, cnt, (kt + 1) * BK, tid);
            stage_b(lBt[cur ^ 1], W2t, ebase, kI, n0, (kt + 1) * BK, tid);
        }
        v16bf a = load_a_frag(lA[cur], lane, wm);
#pragma unroll
        for (int t = 0; t < 4; ++t) {
            v16bf b = load_b_frag(lBt[cur], lane, wn * 64 + t * 16 + (lane & 15));
            acc[t] = __builtin_amdgcn_wmma_f32_16x16x32_bf16(false, a, false, b,
                                                             (short)0, acc[t], false, false);
        }
        wait_async0();
        __syncthreads();
    }

    const int mhi = (lane & 16) ? 8 : 0;
    const int nlow = lane & 15;
#pragma unroll
    for (int r = 0; r < 8; ++r) {
        int mg = m0 + wm * 16 + r + mhi;
        if (mg < cnt) {
            int tok = pair_token[base + mg];
            float wgt = pair_weight[base + mg];
#pragma unroll
            for (int t = 0; t < 4; ++t) {
                int nc = n0 + wn * 64 + t * 16 + nlow;
                float v = (acc[t][r] + b2[(size_t)e * kH + nc]) * wgt;
                atomicAdd(&out[(size_t)tok * kH + nc], v);
            }
        }
    }
}

// ================= host launch =================
extern "C" void kernel_launch(void* const* d_in, const int* in_sizes, int n_in,
                              void* d_out, int out_size, void* d_ws, size_t ws_size,
                              hipStream_t stream) {
    const float* x  = (const float*)d_in[0];   // [2,2048,H]
    const float* Wr = (const float*)d_in[1];   // [H,E]
    const float* br = (const float*)d_in[2];   // [E]
    const float* W1 = (const float*)d_in[3];   // [E,H,I]
    const float* b1 = (const float*)d_in[4];   // [E,I]
    const float* W2 = (const float*)d_in[5];   // [E,I,H]
    const float* b2 = (const float*)d_in[6];   // [E,H]
    float* out = (float*)d_out;                // [T*H] out  +  [1] aux_loss

    // ---- workspace carve-up (~200.5 MB) ----
    char* w = (char*)d_ws;
    __bf16* Xbf  = (__bf16*)w;  w += (size_t)kT * kH * sizeof(__bf16);        // 8 MB
    __bf16* Hbuf = (__bf16*)w;  w += (size_t)kPairs * kI * sizeof(__bf16);    // 64 MB
    __bf16* W1t  = (__bf16*)w;  w += (size_t)kE * kI * kH * sizeof(__bf16);   // 64 MB
    __bf16* W2t  = (__bf16*)w;  w += (size_t)kE * kH * kI * sizeof(__bf16);   // 64 MB
    int*   pair_token  = (int*)w;   w += (size_t)kPairs * sizeof(int);
    float* pair_weight = (float*)w; w += (size_t)kPairs * sizeof(float);
    int*   top2i = (int*)w;         w += (size_t)kT * 2 * sizeof(int);
    float* top2w = (float*)w;       w += (size_t)kT * 2 * sizeof(float);
    int*   meta  = (int*)w;         w += 25 * sizeof(int);   // counts[8] cursor[8] offsets[9]
    float* probs_sum = (float*)w;
    int* counts  = meta;
    int* cursor  = meta + 8;
    int* offsets = meta + 16;

    const int nOut = kT * kH;

    k_zero<<<dim3((nOut + 255) / 256), dim3(256), 0, stream>>>(out, nOut, meta, probs_sum);
    k_cvt<<<dim3((nOut + 255) / 256), dim3(256), 0, stream>>>(x, Xbf, nOut);
    // W1[e]: K=H rows, N=I cols -> W1t[e][I][H];  W2[e]: K=I rows, N=H cols -> W2t[e][H][I]
    k_transpose<<<dim3(kI / 32, kH / 32, kE), dim3(32, 8), 0, stream>>>(W1, W1t, kH, kI);
    k_transpose<<<dim3(kH / 32, kI / 32, kE), dim3(32, 8), 0, stream>>>(W2, W2t, kI, kH);
    k_router<<<dim3(kT / 256), dim3(256), 0, stream>>>(x, Wr, br, counts, probs_sum, top2i, top2w);
    k_finalize<<<dim3(1), dim3(1), 0, stream>>>(counts, probs_sum, offsets, out + (size_t)nOut);
    k_fill<<<dim3(kT / 256), dim3(256), 0, stream>>>(top2i, top2w, offsets, cursor, pair_token, pair_weight);
    k_gemm1<<<dim3(kI / BN, kT / BM, kE), dim3(256), 0, stream>>>(Xbf, W1t, b1, pair_token, counts, offsets, Hbuf);
    k_gemm2<<<dim3(kH / BN, kT / BM, kE), dim3(256), 0, stream>>>(Hbuf, W2t, b2, pair_token, pair_weight, counts, offsets, out);
}
